// GAT_38663295598780
// MI455X (gfx1250) — matrix operand order repaired and checked by reading, not compiled, not run on hardware
//
#include <hip/hip_runtime.h>
#include <hip/hip_bf16.h>

// ---------------------------------------------------------------------------
// GAT (3x GATConv) + 2-layer BiLSTM + pooled MLP head for MI455X (gfx1250).
// GEMMs: v_wmma_f32_16x16x32_f16, double-buffered LDS tiles stored as f16 in
// WMMA operand order; b128 LDS traffic both directions; f32->f16 cvt at stage
// time. Edge phase: atomics. LSTM scan: L2-resident Whh streamed as float4.
// ---------------------------------------------------------------------------

typedef __attribute__((ext_vector_type(16))) _Float16 v16h;
typedef __attribute__((ext_vector_type(8)))  _Float16 v8h;
typedef __attribute__((ext_vector_type(8)))  float    v8f;

#define GAT_N   8192
#define GAT_E   262144
#define GAT_EP  (GAT_E + GAT_N)   // with self loops
#define NEG_SLOPE 0.2f
#define BN_EPS   1e-5f
#define HSTR 40                   // LDS row stride in halves (80B, conflict-free)

// ------------------------------ utilities ----------------------------------

__global__ void fill_f32(float* __restrict__ p, float v, size_t n) {
  size_t i = (size_t)blockIdx.x * blockDim.x + threadIdx.x;
  if (i < n) p[i] = v;
}

__global__ void fill_u32(unsigned* __restrict__ p, unsigned v, size_t n) {
  size_t i = (size_t)blockIdx.x * blockDim.x + threadIdx.x;
  if (i < n) p[i] = v;
}

__global__ void vec_add(const float* __restrict__ a, const float* __restrict__ b,
                        float* __restrict__ o, int n) {
  int i = blockIdx.x * blockDim.x + threadIdx.x;
  if (i < n) o[i] = a[i] + b[i];
}

__device__ inline v8h cvt8(float4 lo, float4 hi) {
  v8h r;
  r[0] = (_Float16)lo.x; r[1] = (_Float16)lo.y;
  r[2] = (_Float16)lo.z; r[3] = (_Float16)lo.w;
  r[4] = (_Float16)hi.x; r[5] = (_Float16)hi.y;
  r[6] = (_Float16)hi.z; r[7] = (_Float16)hi.w;
  return r;
}

// ------------------------ WMMA GEMM (f16 in, f32 acc) ----------------------
// C[M,N] = A[M,K] @ B[K,N] (+bias, optional relu).
// BT==1: B stored [N,K] row-major (torch Linear weight layout).
// Block = 128 threads = 4 waves. Block tile 64x32, K-step 32.
// Each wave: one 16-row A operand, two 16x16 accumulators (2 wmma / K-step).
// Double-buffered LDS; tiles staged through registers (8 floats -> one
// ds_store_b128 of packed halves).

template <int BT>
__global__ __launch_bounds__(128)
void gemm_wmma_f16(const float* __restrict__ A, const float* __restrict__ B,
                   const float* __restrict__ bias, float* __restrict__ C,
                   int M, int N, int K, int doRelu) {
  __shared__ __align__(16) _Float16 AsLds[2][64 * HSTR];  // rows 0..63, k 0..31
  __shared__ __align__(16) _Float16 BsLds[2][32 * HSTR];  // cols 0..31, k 0..31

  const int tid    = threadIdx.x;       // 0..127
  const int wave   = tid >> 5;          // 0..3
  const int lane   = tid & 31;
  const int laneHi = (lane >> 4) & 1;
  const int r0 = blockIdx.x * 64;
  const int c0 = blockIdx.y * 32;

  // WMMA operand addressing
  const int mA  = (wave << 4) + (lane & 15);
  const int kcA = laneHi ? 16 : 0;      // + second chunk at kcA+8
  const int nB  = lane & 15;
  const int kB  = laneHi ? 16 : 0;

  // staging indices: 8-float chunks
  const int aRow0 = tid >> 2,        aC8 = (tid & 3) << 3;   // A chunk 0 (+32 rows for chunk 1)
  const int bN    = tid >> 2,        bC8 = (tid & 3) << 3;   // B chunks (BT path)
  const int gbn   = tid & 31,        gbk = (tid >> 5) << 3;  // B gather (non-BT path)

  float4 aReg[4];   // 2 chunks x 2 float4
  float4 bReg[2];   // BT: 1 chunk x 2 float4
  float  bGat[8];   // non-BT: gathered column

  auto loadTile = [&](int k0) {
#pragma unroll
    for (int i = 0; i < 2; ++i) {       // A: 64x32 f32, 2 chunks of 8 per thread
      const float* p = A + (size_t)(r0 + aRow0 + 32 * i) * K + k0 + aC8;
      aReg[2 * i + 0] = *(const float4*)p;
      aReg[2 * i + 1] = *(const float4*)(p + 4);
    }
    if (BT) {                           // B: 32 rows (cols of C) x 32 k
      const float* p = B + (size_t)(c0 + bN) * K + k0 + bC8;
      bReg[0] = *(const float4*)p;
      bReg[1] = *(const float4*)(p + 4);
    } else {                            // gathered transpose: col gbn, k gbk..+7
#pragma unroll
      for (int j = 0; j < 8; ++j)
        bGat[j] = B[(size_t)(k0 + gbk + j) * N + c0 + gbn];
    }
  };

  auto storeTile = [&](int buf) {
#pragma unroll
    for (int i = 0; i < 2; ++i)
      *(v8h*)(&AsLds[buf][(aRow0 + 32 * i) * HSTR + aC8]) =
          cvt8(aReg[2 * i + 0], aReg[2 * i + 1]);
    if (BT) {
      *(v8h*)(&BsLds[buf][bN * HSTR + bC8]) = cvt8(bReg[0], bReg[1]);
    } else {
      v8h h;
#pragma unroll
      for (int j = 0; j < 8; ++j) h[j] = (_Float16)bGat[j];
      *(v8h*)(&BsLds[buf][gbn * HSTR + gbk]) = h;
    }
  };

  v8f acc0 = {}, acc1 = {};

  loadTile(0);
  storeTile(0);
  __syncthreads();

  const int nT = K >> 5;
  for (int t = 0; t < nT; ++t) {
    const int cur = t & 1;
    if (t + 1 < nT) loadTile((t + 1) << 5);
    if (t + 2 < nT)  // emits global_prefetch_b8 for tile t+2
      __builtin_prefetch(A + (size_t)(r0 + (tid >> 1)) * K + ((t + 2) << 5), 0, 1);

    // A operand: two ds_load_b128 per lane
    const _Float16* ab = &AsLds[cur][mA * HSTR + kcA];
    v8h alo = *(const v8h*)ab;
    v8h ahi = *(const v8h*)(ab + 8);
    v16h a;
#pragma unroll
    for (int j = 0; j < 8; ++j) { a[j] = alo[j]; a[8 + j] = ahi[j]; }

    // B operands for the two 16-col sub-tiles
    const _Float16* bp0 = &BsLds[cur][nB * HSTR + kB];
    const _Float16* bp1 = &BsLds[cur][(16 + nB) * HSTR + kB];
    v8h bl0 = *(const v8h*)bp0, bh0 = *(const v8h*)(bp0 + 8);
    v8h bl1 = *(const v8h*)bp1, bh1 = *(const v8h*)(bp1 + 8);
    v16h b0, b1;
#pragma unroll
    for (int j = 0; j < 8; ++j) {
      b0[j] = bl0[j]; b0[8 + j] = bh0[j];
      b1[j] = bl1[j]; b1[8 + j] = bh1[j];
    }

    acc0 = __builtin_amdgcn_wmma_f32_16x16x32_f16(false, a, false, b0, (short)0,
                                                  acc0, false, false);
    acc1 = __builtin_amdgcn_wmma_f32_16x16x32_f16(false, a, false, b1, (short)0,
                                                  acc1, false, false);

    if (t + 1 < nT) storeTile((t + 1) & 1);
    __syncthreads();
  }

  // D layout: lane -> col lane%16; VGPR r -> row r (+8 for lanes 16-31)
  const int rowBase = r0 + (wave << 4) + (laneHi << 3);
#pragma unroll
  for (int r = 0; r < 8; ++r) {
    int row = rowBase + r;
    int colA = c0 + nB, colB = c0 + 16 + nB;
    float va = acc0[r], vb = acc1[r];
    if (bias) { va += bias[colA]; vb += bias[colB]; }
    if (doRelu) { va = fmaxf(va, 0.f); vb = fmaxf(vb, 0.f); }
    C[(size_t)row * N + colA] = va;
    C[(size_t)row * N + colB] = vb;
  }
}

// --------------------------- GAT edge phase --------------------------------

__global__ void node_scores(const float* __restrict__ h,
                            const float* __restrict__ a_s,
                            const float* __restrict__ a_d,
                            float* __restrict__ s_src, float* __restrict__ s_dst,
                            int n, int F) {
  int i = blockIdx.x * blockDim.x + threadIdx.x;
  if (i >= n) return;
  const float4* row = (const float4*)(h + (size_t)i * F);
  const float4* as4 = (const float4*)a_s;
  const float4* ad4 = (const float4*)a_d;
  float ss = 0.f, sd = 0.f;
  for (int f = 0; f < (F >> 2); ++f) {
    float4 v = row[f], s4 = as4[f], d4 = ad4[f];
    ss += v.x * s4.x + v.y * s4.y + v.z * s4.z + v.w * s4.w;
    sd += v.x * d4.x + v.y * d4.y + v.z * d4.z + v.w * d4.w;
  }
  s_src[i] = ss; s_dst[i] = sd;
}

__device__ inline unsigned f32_ord_encode(float f) {
  unsigned u = __float_as_uint(f);
  return (u >> 31) ? ~u : (u | 0x80000000u);
}
__device__ inline float f32_ord_decode(unsigned u) {
  return (u & 0x80000000u) ? __uint_as_float(u & 0x7FFFFFFFu)
                           : __uint_as_float(~u);
}

__global__ void edge_logits(const int* __restrict__ src, const int* __restrict__ dst,
                            const float* __restrict__ s_src, const float* __restrict__ s_dst,
                            int E, int n, float* __restrict__ e_val,
                            unsigned* __restrict__ mmax) {
  int e = blockIdx.x * blockDim.x + threadIdx.x;
  if (e >= E + n) return;
  int s, d;
  if (e < E) { s = src[e]; d = dst[e]; } else { s = d = e - E; }
  float v = s_src[s] + s_dst[d];
  v = (v >= 0.f) ? v : NEG_SLOPE * v;
  e_val[e] = v;
  atomicMax(&mmax[d], f32_ord_encode(v));
}

__global__ void edge_exp(const float* __restrict__ e_val,
                         const unsigned* __restrict__ mmax,
                         const int* __restrict__ dst, int E, int n,
                         float* __restrict__ exv, float* __restrict__ ssum) {
  int e = blockIdx.x * blockDim.x + threadIdx.x;
  if (e >= E + n) return;
  int d = (e < E) ? dst[e] : (e - E);
  float ex = __expf(e_val[e] - f32_ord_decode(mmax[d]));
  exv[e] = ex;
  atomicAdd(&ssum[d], ex);
}

__global__ void edge_scatter(const float* __restrict__ exv,
                             const float* __restrict__ ssum,
                             const int* __restrict__ src, const int* __restrict__ dst,
                             int E, int n, const float* __restrict__ h, int F,
                             float* __restrict__ agg, float* __restrict__ alpha_out) {
  int e = blockIdx.x;           // one block per edge, blockDim.x == F
  int f = threadIdx.x;
  int s, d;
  if (e < E) { s = src[e]; d = dst[e]; } else { s = d = e - E; }
  float alpha = exv[e] / (ssum[d] + 1e-16f);
  if (f == 0) alpha_out[e] = alpha;
  atomicAdd(&agg[(size_t)d * F + f], alpha * h[(size_t)s * F + f]);
}

__global__ void bias_relu(const float* __restrict__ agg, const float* __restrict__ b,
                          float* __restrict__ out, int n, int F) {
  size_t i = (size_t)blockIdx.x * blockDim.x + threadIdx.x;
  if (i >= (size_t)n * F) return;
  float v = agg[i] + b[i % F];
  out[i] = fmaxf(v, 0.f);
}

__global__ void write_edges(const int* __restrict__ src, const int* __restrict__ dst,
                            int E, int n, float* __restrict__ out) {
  int e = blockIdx.x * blockDim.x + threadIdx.x;
  int EP = E + n;
  if (e >= EP) return;
  int s, d;
  if (e < E) { s = src[e]; d = dst[e]; } else { s = d = e - E; }
  out[e] = (float)s;
  out[EP + e] = (float)d;
}

// ------------------------------- LSTM scan ---------------------------------
// One workgroup (512 threads) per direction. xp precomputed by WMMA GEMM.
// Whh (256 KB) is resident in WGP$/L2 (192 MB) across all 8192 steps and
// streamed as float4; h and gates staged in LDS.

__global__ __launch_bounds__(512)
void lstm_scan(const float* __restrict__ xp,  // [T, 512]
               const float* __restrict__ Whh, // [512, 128]
               float* __restrict__ y,         // [T, yStride]
               int T, int yStride, int yOff, int reverse) {
  __shared__ __align__(16) float hb[128];
  __shared__ float gb[512];
  const int tid = threadIdx.x;       // one gate element per thread
  const float4* wrow = (const float4*)(Whh + (size_t)tid * 128);
  const float4* hb4 = (const float4*)hb;
  float c = 0.f;
  if (tid < 128) hb[tid] = 0.f;
  __syncthreads();

  for (int step = 0; step < T; ++step) {
    int t = reverse ? (T - 1 - step) : step;
    float g = xp[(size_t)t * 512 + tid];
#pragma unroll 8
    for (int k = 0; k < 32; ++k) {
      float4 w = wrow[k], h4 = hb4[k];
      g += w.x * h4.x + w.y * h4.y + w.z * h4.z + w.w * h4.w;
    }
    gb[tid] = g;
    __syncthreads();
    if (tid < 128) {
      float ig = 1.f / (1.f + __expf(-gb[tid]));
      float fg = 1.f / (1.f + __expf(-gb[128 + tid]));
      float gg = tanhf(gb[256 + tid]);
      float og = 1.f / (1.f + __expf(-gb[384 + tid]));
      c = fg * c + ig * gg;
      float hnew = og * tanhf(c);
      hb[tid] = hnew;
      y[(size_t)t * yStride + yOff + tid] = hnew;
    }
    __syncthreads();
  }
}

// ---------------------- pool + MLP head (fused) ----------------------------

__global__ __launch_bounds__(256)
void pool_head(const float* __restrict__ y,      // [N, 256]
               const float* __restrict__ fc1w, const float* __restrict__ fc1b,
               const float* __restrict__ g1,   const float* __restrict__ be1,
               const float* __restrict__ fc2w, const float* __restrict__ fc2b,
               const float* __restrict__ g2,   const float* __restrict__ be2,
               const float* __restrict__ fc3w, const float* __restrict__ fc3b,
               float* __restrict__ logits) {
  __shared__ float pooled[256];
  __shared__ float z1[256];
  __shared__ float z2[64];
  const int b = blockIdx.x, tid = threadIdx.x;
  const float bnScale = 1.f / sqrtf(1.f + BN_EPS);

  float s = 0.f;
  for (int r = 0; r < 16; ++r)
    s += y[((size_t)b * 16 + r) * 256 + tid];
  pooled[tid] = s * (1.f / 16.f);
  __syncthreads();

  float z = fc1b[tid];
  for (int k = 0; k < 256; ++k) z += pooled[k] * fc1w[(size_t)tid * 256 + k];
  z = fmaxf(z, 0.f);
  z1[tid] = z * (g1[tid] * bnScale) + be1[tid];
  __syncthreads();

  if (tid < 64) {
    float z_ = fc2b[tid];
    for (int k = 0; k < 256; ++k) z_ += z1[k] * fc2w[(size_t)tid * 256 + k];
    z_ = fmaxf(z_, 0.f);
    z2[tid] = z_ * (g2[tid] * bnScale) + be2[tid];
  }
  __syncthreads();

  if (tid < 3) {
    float z_ = fc3b[tid];
    for (int k = 0; k < 64; ++k) z_ += z2[k] * fc3w[(size_t)tid * 64 + k];
    logits[(size_t)b * 3 + tid] = z_;
  }
}

// ------------------------------- launcher ----------------------------------

extern "C" void kernel_launch(void* const* d_in, const int* in_sizes, int n_in,
                              void* d_out, int out_size, void* d_ws, size_t ws_size,
                              hipStream_t stream) {
  (void)in_sizes; (void)n_in; (void)out_size; (void)ws_size;
  const int Nn = GAT_N, E = GAT_E, EP = GAT_EP;

  int k = 0;
  const float* x    = (const float*)d_in[k++];
  const int*   ei   = (const int*)d_in[k++];
  const float* W1   = (const float*)d_in[k++];
  const float* a1s  = (const float*)d_in[k++];
  const float* a1d  = (const float*)d_in[k++];
  const float* b1   = (const float*)d_in[k++];
  const float* W2   = (const float*)d_in[k++];
  const float* a2s  = (const float*)d_in[k++];
  const float* a2d  = (const float*)d_in[k++];
  const float* b2   = (const float*)d_in[k++];
  const float* W3   = (const float*)d_in[k++];
  const float* a3s  = (const float*)d_in[k++];
  const float* a3d  = (const float*)d_in[k++];
  const float* b3   = (const float*)d_in[k++];
  const float* Wih0  = (const float*)d_in[k++];
  const float* Whh0  = (const float*)d_in[k++];
  const float* bih0  = (const float*)d_in[k++];
  const float* bhh0  = (const float*)d_in[k++];
  const float* Wih0r = (const float*)d_in[k++];
  const float* Whh0r = (const float*)d_in[k++];
  const float* bih0r = (const float*)d_in[k++];
  const float* bhh0r = (const float*)d_in[k++];
  const float* Wih1  = (const float*)d_in[k++];
  const float* Whh1  = (const float*)d_in[k++];
  const float* bih1  = (const float*)d_in[k++];
  const float* bhh1  = (const float*)d_in[k++];
  const float* Wih1r = (const float*)d_in[k++];
  const float* Whh1r = (const float*)d_in[k++];
  const float* bih1r = (const float*)d_in[k++];
  const float* bhh1r = (const float*)d_in[k++];
  const float* fc1w = (const float*)d_in[k++];
  const float* fc1b = (const float*)d_in[k++];
  const float* g1   = (const float*)d_in[k++];
  const float* be1  = (const float*)d_in[k++];
  const float* fc2w = (const float*)d_in[k++];
  const float* fc2b = (const float*)d_in[k++];
  const float* g2   = (const float*)d_in[k++];
  const float* be2  = (const float*)d_in[k++];
  const float* fc3w = (const float*)d_in[k++];
  const float* fc3b = (const float*)d_in[k++];

  const int* src = ei;       // edge_index[0]
  const int* dst = ei + E;   // edge_index[1]

  // workspace carve-up
  float* ws = (float*)d_ws;
  size_t off = 0;
  auto wsAlloc = [&](size_t n) { float* p = ws + off; off += n; return p; };
  float* hpre1 = wsAlloc((size_t)Nn * 128);
  float* hpre2 = wsAlloc((size_t)Nn * 256);
  float* hpre3 = wsAlloc((size_t)Nn * 128);
  float* agg   = wsAlloc((size_t)Nn * 256);
  float* sSrc  = wsAlloc(Nn);
  float* sDst  = wsAlloc(Nn);
  float* eVal  = wsAlloc(EP);
  float* exv   = wsAlloc(EP);
  float* mbuf  = wsAlloc(Nn);        // used as unsigned
  float* ssum  = wsAlloc(Nn);
  float* xpF   = wsAlloc((size_t)Nn * 512);
  float* xpB   = wsAlloc((size_t)Nn * 512);
  float* y0    = wsAlloc((size_t)Nn * 256);
  float* y1    = wsAlloc((size_t)Nn * 256);
  float* bsum  = wsAlloc(4 * 512);

  // d_out layout: logits[512*3] | a1[EP] | a2[EP] | a3[EP] | src[EP] | dst[EP]
  float* out       = (float*)d_out;
  float* outLogits = out;
  float* outA1     = out + 512 * 3;
  float* outA2     = outA1 + EP;
  float* outA3     = outA2 + EP;
  float* outEdges  = outA3 + EP;

  auto cdiv = [](size_t a, size_t b) { return (unsigned)((a + b - 1) / b); };

  auto run_gat = [&](const float* xin, int Fin, int Fout, const float* W,
                     const float* as_, const float* ad_, const float* bb,
                     float* hpre, float* alphaOut) {
    gemm_wmma_f16<0><<<dim3(Nn / 64, Fout / 32), 128, 0, stream>>>(
        xin, W, nullptr, hpre, Nn, Fout, Fin, /*relu=*/0);
    node_scores<<<cdiv(Nn, 256), 256, 0, stream>>>(hpre, as_, ad_, sSrc, sDst, Nn, Fout);
    fill_u32<<<cdiv(Nn, 256), 256, 0, stream>>>((unsigned*)mbuf, 0u, Nn);
    fill_f32<<<cdiv(Nn, 256), 256, 0, stream>>>(ssum, 0.f, Nn);
    fill_f32<<<cdiv((size_t)Nn * Fout, 256), 256, 0, stream>>>(agg, 0.f, (size_t)Nn * Fout);
    edge_logits<<<cdiv(EP, 256), 256, 0, stream>>>(src, dst, sSrc, sDst, E, Nn,
                                                   eVal, (unsigned*)mbuf);
    edge_exp<<<cdiv(EP, 256), 256, 0, stream>>>(eVal, (unsigned*)mbuf, dst, E, Nn,
                                                exv, ssum);
    edge_scatter<<<EP, Fout, 0, stream>>>(exv, ssum, src, dst, E, Nn, hpre, Fout,
                                          agg, alphaOut);
    bias_relu<<<cdiv((size_t)Nn * Fout, 256), 256, 0, stream>>>(agg, bb, hpre, Nn, Fout);
  };

  // ---- GAT stack ----
  run_gat(x,     1024, 128, W1, a1s, a1d, b1, hpre1, outA1);
  run_gat(hpre1,  128, 256, W2, a2s, a2d, b2, hpre2, outA2);
  run_gat(hpre2,  256, 128, W3, a3s, a3d, b3, hpre3, outA3);

  // ---- BiLSTM layer 0 (input hpre3 [N,128]) ----
  vec_add<<<2, 256, 0, stream>>>(bih0,  bhh0,  bsum + 0 * 512, 512);
  vec_add<<<2, 256, 0, stream>>>(bih0r, bhh0r, bsum + 1 * 512, 512);
  gemm_wmma_f16<1><<<dim3(Nn / 64, 512 / 32), 128, 0, stream>>>(
      hpre3, Wih0, bsum + 0 * 512, xpF, Nn, 512, 128, 0);
  gemm_wmma_f16<1><<<dim3(Nn / 64, 512 / 32), 128, 0, stream>>>(
      hpre3, Wih0r, bsum + 1 * 512, xpB, Nn, 512, 128, 0);
  lstm_scan<<<1, 512, 0, stream>>>(xpF, Whh0,  y0, Nn, 256, 0,   /*rev=*/0);
  lstm_scan<<<1, 512, 0, stream>>>(xpB, Whh0r, y0, Nn, 256, 128, /*rev=*/1);

  // ---- BiLSTM layer 1 (input y0 [N,256]) ----
  vec_add<<<2, 256, 0, stream>>>(bih1,  bhh1,  bsum + 2 * 512, 512);
  vec_add<<<2, 256, 0, stream>>>(bih1r, bhh1r, bsum + 3 * 512, 512);
  gemm_wmma_f16<1><<<dim3(Nn / 64, 512 / 32), 128, 0, stream>>>(
      y0, Wih1, bsum + 2 * 512, xpF, Nn, 512, 256, 0);
  gemm_wmma_f16<1><<<dim3(Nn / 64, 512 / 32), 128, 0, stream>>>(
      y0, Wih1r, bsum + 3 * 512, xpB, Nn, 512, 256, 0);
  lstm_scan<<<1, 512, 0, stream>>>(xpF, Whh1,  y1, Nn, 256, 0,   /*rev=*/0);
  lstm_scan<<<1, 512, 0, stream>>>(xpB, Whh1r, y1, Nn, 256, 128, /*rev=*/1);

  // ---- pool + MLP head ----
  pool_head<<<Nn / 16, 256, 0, stream>>>(y1, fc1w, fc1b, g1, be1,
                                         fc2w, fc2b, g2, be2, fc3w, fc3b,
                                         outLogits);

  // ---- stacked (src, dst) with self loops ----
  write_edges<<<cdiv(EP, 256), 256, 0, stream>>>(src, dst, E, Nn, outEdges);
}
